// ConvGRU_7318624272932
// MI455X (gfx1250) — matrix operand order repaired and checked by reading, not compiled, not run on hardware
//
#include <hip/hip_runtime.h>

#define DIMC   128
#define K2     256
#define HWSZ   4096
#define TSTEPS 16
#define NTILE  128
#define NSPAT  8192

typedef __attribute__((ext_vector_type(16))) __bf16          v16bf;
typedef __attribute__((ext_vector_type(8)))  float           v8f;
typedef __attribute__((ext_vector_type(16))) unsigned short  us16;
typedef __attribute__((ext_vector_type(8)))  unsigned short  us8;
typedef __attribute__((ext_vector_type(4)))  unsigned int    u32x4;
typedef __attribute__((ext_vector_type(8)))  int             i32x8;
typedef __attribute__((ext_vector_type(4)))  int             i32x4;

__device__ __forceinline__ unsigned short f2bf(float f) {
    unsigned int u = __float_as_uint(f);
    unsigned int r = (u + 0x7FFFu + ((u >> 16) & 1u)) >> 16;  // round-to-nearest-even
    return (unsigned short)r;
}
__device__ __forceinline__ float bf2f(unsigned short b) {
    return __uint_as_float(((unsigned int)b) << 16);
}
__device__ __forceinline__ float sigm(float x) { return 1.0f / (1.0f + __expf(-x)); }

// ---- TDM: DMA one 64KB bf16 weight panel (32768 elems) global -> LDS ----
// D# per ISA ch.8: 1-D tensor, tensor_dim0 = tile_dim0 = 32768, data_size = 2B.
__device__ __forceinline__ void tdm_load_w(const unsigned short* gptr, unsigned lds_off) {
    unsigned long long ga = (unsigned long long)(size_t)gptr;
    u32x4 g0;
    g0[0] = 1u;                                                // count=1, user descriptor
    g0[1] = lds_off;                                           // lds_addr (bytes)
    g0[2] = (unsigned)(ga & 0xFFFFFFFFu);                      // global_addr[31:0]
    g0[3] = (unsigned)((ga >> 32) & 0x01FFFFFFu) | 0x80000000u;// global_addr[56:32] | type=2
    i32x8 g1;
    g1[0] = 0x00010000;        // workgroup_mask=0, data_size=1 (2 bytes)
    g1[1] = (int)0x80000000;   // tensor_dim0 lo16 = 0x8000 (32768)
    g1[2] = 0x00010000;        // tensor_dim0 hi16 = 0, tensor_dim1 lo16 = 1
    g1[3] = (int)0x80000000;   // tensor_dim1 hi16 = 0, tile_dim0 = 0x8000
    g1[4] = 0x00000001;        // tile_dim1 = 1, tile_dim2 = 0
    g1[5] = 0x00008000;        // tensor_dim0_stride = 32768 (lo32)
    g1[6] = 0;                 // stride0 hi16 = 0, tensor_dim1_stride lo16 = 0
    g1[7] = 0;
    i32x4 gz = {0, 0, 0, 0};
#if defined(__clang_major__) && (__clang_major__ >= 23)
    i32x8 gz8 = {0, 0, 0, 0, 0, 0, 0, 0};
    __builtin_amdgcn_tensor_load_to_lds(g0, g1, gz, gz, gz8, 0);
#else
    __builtin_amdgcn_tensor_load_to_lds(g0, g1, gz, gz, 0);
#endif
}

// One wave computes D[mbase:mbase+16, 0:128] += W[mbase:+16, 0:256] * X[0:256, 0:128].
template<bool USE_RH>
__device__ __forceinline__ void gemm_tile(const unsigned short* __restrict__ w,
                                          const unsigned short* __restrict__ xh,
                                          const unsigned short* __restrict__ rh,
                                          int mbase, int lane, v8f* acc) {
    const int l15 = lane & 15;
    const int hi  = (lane >> 4) & 1;
    const int row = mbase + l15;
#pragma unroll
    for (int k = 0; k < 8; ++k) {
        const int kbase = k * 32;
        // A fragment (16x32 bf16): lane holds row M, 16 contiguous K at K0
        const int K0 = kbase + hi * 16;
        us8 alo = *(const us8*)(w + row * K2 + K0);
        us8 ahi = *(const us8*)(w + row * K2 + K0 + 8);
        us16 ap;
#pragma unroll
        for (int j = 0; j < 8; ++j) { ap[j] = alo[j]; ap[j + 8] = ahi[j]; }
        v16bf a = __builtin_bit_cast(v16bf, ap);
        // B fragments (32x16 bf16): lane holds col N, 16 contiguous K at Kb
        const int Kb = kbase + hi * 16;
#pragma unroll
        for (int n = 0; n < 8; ++n) {
            const int col = n * 16 + l15;
            const unsigned short* bp;
            if (USE_RH && kbase >= 128) bp = rh + col * DIMC + (Kb - 128);
            else                        bp = xh + col * K2  + Kb;
            v16bf bfrag = __builtin_bit_cast(v16bf, *(const us16*)bp);
            acc[n] = __builtin_amdgcn_wmma_f32_16x16x32_bf16(
                false, a, false, bfrag, (short)0, acc[n], false, false);
        }
    }
}

// ---- prep: convert all three weight matrices f32 -> bf16 once per launch ----
__global__ __launch_bounds__(256) void convert_weights(
    const float* __restrict__ Wz, const float* __restrict__ Wr,
    const float* __restrict__ Wh, unsigned short* __restrict__ wbf)
{
    int i = blockIdx.x * 256 + threadIdx.x;        // 0..24575, one float4 each
    int mat  = i >> 13;                            // 0=z, 1=r, 2=h
    int idx4 = i & 8191;
    const float* src = (mat == 0) ? Wz : (mat == 1) ? Wr : Wh;
    float4 wv = ((const float4*)src)[idx4];
    unsigned int lo = (unsigned int)f2bf(wv.x) | ((unsigned int)f2bf(wv.y) << 16);
    unsigned int hi = (unsigned int)f2bf(wv.z) | ((unsigned int)f2bf(wv.w) << 16);
    *(uint2*)(wbf + (size_t)mat * 32768 + (size_t)idx4 * 4) = make_uint2(lo, hi);
}

__global__ __launch_bounds__(256, 1) void convgru_step(
    const float* __restrict__ video, float* __restrict__ out,
    const unsigned short* __restrict__ wbf,
    const float* __restrict__ bz, const float* __restrict__ br,
    const float* __restrict__ bh, int t)
{
    extern __shared__ unsigned short smem[];
    unsigned short* ldsw0 = smem;             // weight buffer 0 (64 KB)
    unsigned short* ldsw1 = smem + 32768;     // weight buffer 1 (64 KB)
    unsigned short* ldsxh = smem + 65536;     // [128 n][256 c] bf16, c<128=x, c>=128=h (64 KB)
    unsigned short* ldsrh = smem + 98304;     // [128 n][128 c] bf16 r*h (32 KB)

    const int tid   = threadIdx.x;
    const int lane  = tid & 31;
    const int wid   = tid >> 5;
    const int sbase = blockIdx.x * NTILE;
    const int b     = sbase / HWSZ;
    const int hw0   = sbase % HWSZ;

    const float* xg = video + (size_t)(b * TSTEPS + t) * DIMC * HWSZ + hw0;
    const float* hg = (t > 0) ? out + (size_t)(b * TSTEPS + t - 1) * DIMC * HWSZ + hw0 : out;
    float*       og = out   + (size_t)(b * TSTEPS + t) * DIMC * HWSZ + hw0;

    // kick the r-gate weight DMA before staging the activation tile
    if (wid == 0) tdm_load_w(wbf + 1 * 32768, 0u);               // Wr -> W0

    // ---- stage x|h tile as bf16, [n][c] with c contiguous ----
#pragma unroll
    for (int it = 0; it < 16; ++it) {
        int i  = tid + it * 256;              // 4096 items: 128 ch x 32 float4-cols
        int c  = i >> 5;
        int n4 = (i & 31) * 4;
        float4 xv = *(const float4*)(xg + (size_t)c * HWSZ + n4);
        ldsxh[(n4 + 0) * K2 + c] = f2bf(xv.x);
        ldsxh[(n4 + 1) * K2 + c] = f2bf(xv.y);
        ldsxh[(n4 + 2) * K2 + c] = f2bf(xv.z);
        ldsxh[(n4 + 3) * K2 + c] = f2bf(xv.w);
        float4 hv = make_float4(0.f, 0.f, 0.f, 0.f);
        if (t > 0) hv = *(const float4*)(hg + (size_t)c * HWSZ + n4);
        ldsxh[(n4 + 0) * K2 + DIMC + c] = f2bf(hv.x);
        ldsxh[(n4 + 1) * K2 + DIMC + c] = f2bf(hv.y);
        ldsxh[(n4 + 2) * K2 + DIMC + c] = f2bf(hv.z);
        ldsxh[(n4 + 3) * K2 + DIMC + c] = f2bf(hv.w);
    }
    __builtin_amdgcn_s_wait_tensorcnt(0);
    __syncthreads();                                              // W0 + xh ready

    const int mbase = wid * 16;
    const int l15   = lane & 15;
    const int hi8   = ((lane >> 4) & 1) * 8;
    const v8f vzero = {0.f, 0.f, 0.f, 0.f, 0.f, 0.f, 0.f, 0.f};

    v8f acc[8];
    float bias[8];

    // ---- r gate (W0); prefetch Wz -> W1 under the GEMM ----
    if (wid == 0) tdm_load_w(wbf + 0 * 32768, 65536u);            // Wz -> W1
#pragma unroll
    for (int n = 0; n < 8; ++n) acc[n] = vzero;
    gemm_tile<false>(ldsw0, ldsxh, ldsrh, mbase, lane, acc);
#pragma unroll
    for (int v = 0; v < 8; ++v) bias[v] = br[mbase + hi8 + v];
#pragma unroll
    for (int n = 0; n < 8; ++n) {
        const int col = n * 16 + l15;
#pragma unroll
        for (int v = 0; v < 8; ++v) {
            const int M = mbase + hi8 + v;
            float r    = sigm(acc[n][v] + bias[v]);
            float hval = bf2f(ldsxh[col * K2 + DIMC + M]);
            ldsrh[col * DIMC + M] = f2bf(r * hval);
        }
    }
    __builtin_amdgcn_s_wait_tensorcnt(0);
    __syncthreads();                                              // W1 + rh ready, W0 free

    // ---- z gate (W1); prefetch Wh -> W0 under the GEMM ----
    if (wid == 0) tdm_load_w(wbf + 2 * 32768, 0u);                // Wh -> W0
#pragma unroll
    for (int n = 0; n < 8; ++n) acc[n] = vzero;
    gemm_tile<false>(ldsw1, ldsxh, ldsrh, mbase, lane, acc);
#pragma unroll
    for (int v = 0; v < 8; ++v) bias[v] = bz[mbase + hi8 + v];
    v8f zg[8];
#pragma unroll
    for (int n = 0; n < 8; ++n) {
#pragma unroll
        for (int v = 0; v < 8; ++v) zg[n][v] = sigm(acc[n][v] + bias[v]);
    }
    __builtin_amdgcn_s_wait_tensorcnt(0);
    __syncthreads();                                              // W0 (Wh) ready

    // ---- candidate: hc = tanh(Wh.[x; r*h] + bh); h_new = (1-z)h + z*hc ----
#pragma unroll
    for (int n = 0; n < 8; ++n) acc[n] = vzero;
    gemm_tile<true>(ldsw0, ldsxh, ldsrh, mbase, lane, acc);
#pragma unroll
    for (int v = 0; v < 8; ++v) bias[v] = bh[mbase + hi8 + v];
#pragma unroll
    for (int n = 0; n < 8; ++n) {
        const int col = n * 16 + l15;
#pragma unroll
        for (int v = 0; v < 8; ++v) {
            const int M = mbase + hi8 + v;
            float hc   = tanhf(acc[n][v] + bias[v]);
            float hval = bf2f(ldsxh[col * K2 + DIMC + M]);
            float z    = zg[n][v];
            og[(size_t)M * HWSZ + col] = (1.0f - z) * hval + z * hc;
        }
    }
}

extern "C" void kernel_launch(void* const* d_in, const int* in_sizes, int n_in,
                              void* d_out, int out_size, void* d_ws, size_t ws_size,
                              hipStream_t stream) {
    (void)in_sizes; (void)n_in; (void)out_size; (void)ws_size;
    const float* video = (const float*)d_in[0];
    const float* Wz = (const float*)d_in[1];
    const float* bz = (const float*)d_in[2];
    const float* Wr = (const float*)d_in[3];
    const float* br = (const float*)d_in[4];
    const float* Wh = (const float*)d_in[5];
    const float* bh = (const float*)d_in[6];
    float* out = (float*)d_out;
    unsigned short* wbf = (unsigned short*)d_ws;   // 3 * 32768 bf16 = 196608 B

    // one-time (per launch) weight conversion f32 -> bf16 into workspace
    convert_weights<<<dim3(96), dim3(256), 0, stream>>>(Wz, Wr, Wh, wbf);

    dim3 grid(NSPAT / NTILE);   // 64 blocks
    dim3 block(256);            // 8 wave32
    size_t lds_bytes = (size_t)(32768 + 32768 + 32768 + 16384) * sizeof(unsigned short); // 224 KB

    for (int t = 0; t < TSTEPS; ++t) {
        convgru_step<<<grid, block, lds_bytes, stream>>>(video, out, wbf, bz, br, bh, t);
    }
}
